// DynamicAttentionNetwork_38568806318296
// MI455X (gfx1250) — compile-verified, matrix operand order checked by reading, not run on hardware
//
#include <hip/hip_runtime.h>

#define N_   4096
#define B_   2
#define H_   64
#define BN_  (B_ * N_)                 // 8192
#define BNH_ ((size_t)BN_ * H_)        // 524288
#define OUTOFF_ ((size_t)BN_ * 8)      // 65536 floats of out2

typedef __attribute__((ext_vector_type(16))) __bf16       v16bf;
typedef __attribute__((ext_vector_type(8)))  float        v8f;
typedef __attribute__((ext_vector_type(4)))  unsigned int u32x4;

union Frag {
  v16bf v;
  unsigned short u[16];
  u32x4 q[2];
};

// native fptrunc -> v_cvt bf16 (RNE in hardware), 1 VALU op instead of 3
__device__ __forceinline__ unsigned short f2bf(float f) {
  __bf16 h = (__bf16)f;
  return __builtin_bit_cast(unsigned short, h);
}

// K index inside a 32-wide K window for 16-bit A/B fragments (wave32):
// elements 0..7 -> K = 8*half + i ; elements 8..15 -> K = 16 + 8*half + (i-8)
__device__ __forceinline__ int kmap(int i, int hf) {
  return (i & 7) + 8 * hf + ((i & 8) ? 16 : 0);
}

__device__ __forceinline__ v8f wmma_bf16(const Frag& a, const Frag& b, v8f c) {
  return __builtin_amdgcn_wmma_f32_16x16x32_bf16(false, a.v, false, b.v,
                                                 (short)0, c, false, false);
}

__device__ __forceinline__ int edge_pred(int e) {
  return (e == 1 || e == 5) ? 2 : ((e == 3 || e == 4) ? 1 : 0);
}

// ---------------------------------------------------------------------------
// Kernel 1: projections  q_t = h_t @ Wq[t]^T  (p=0..2),  k_e = h_pred @ Wk[e]^T
// grid (128, 9), block 128 (4 waves, each one 16-row tile)
// ---------------------------------------------------------------------------
__global__ __launch_bounds__(128) void proj_kernel(
    const float* __restrict__ h0, const float* __restrict__ h1,
    const float* __restrict__ h2, const float* __restrict__ Wq,
    const float* __restrict__ Wk, float* __restrict__ qf32,
    float* __restrict__ kf32) {
  const int lane = threadIdx.x & 31;
  const int wid  = threadIdx.x >> 5;
  const int hf   = lane >> 4;
  const int ln   = lane & 15;
  const int p    = blockIdx.y;
  const int rowbase = (blockIdx.x * 4 + wid) * 16;   // [0, BN)

  const float* in;
  const float* W;
  float* out;
  if (p < 3) {
    in = (p == 0) ? h0 : (p == 1) ? h1 : h2;
    W  = Wq + (size_t)p * 4096;
    out = qf32 + (size_t)p * BNH_;
  } else {
    int e = p - 3, pr = edge_pred(e);
    in = (pr == 0) ? h0 : (pr == 1) ? h1 : h2;
    W  = Wk + (size_t)e * 4096;
    out = kf32 + (size_t)e * BNH_;
  }

  // A fragments: rows of `in`, converted fp32 -> bf16
  Frag a[2];
  const float* arow = in + (size_t)(rowbase + ln) * 64;
#pragma unroll
  for (int f = 0; f < 2; ++f)
#pragma unroll
    for (int i = 0; i < 16; ++i)
      a[f].u[i] = f2bf(arow[f * 32 + kmap(i, hf)]);

#pragma unroll
  for (int ot = 0; ot < 4; ++ot) {
    v8f c = {0.f, 0.f, 0.f, 0.f, 0.f, 0.f, 0.f, 0.f};
    const float* wrow = W + (size_t)(ot * 16 + ln) * 64;  // B[h,o] = W[o,h]
#pragma unroll
    for (int f = 0; f < 2; ++f) {
      Frag bw;
#pragma unroll
      for (int i = 0; i < 16; ++i)
        bw.u[i] = f2bf(wrow[f * 32 + kmap(i, hf)]);
      c = wmma_bf16(a[f], bw, c);
    }
#pragma unroll
    for (int r = 0; r < 8; ++r) {
      int m = r + 8 * hf;
      out[(size_t)(rowbase + m) * 64 + ot * 16 + ln] = c[r];
    }
  }
}

// ---------------------------------------------------------------------------
// Kernel 2: normalize q/k rows, convert q/k/h to bf16
// grid (BN/8, 12), block 256 (8 waves, one row each)
// ---------------------------------------------------------------------------
__global__ __launch_bounds__(256) void norm_kernel(
    const float* __restrict__ h0, const float* __restrict__ h1,
    const float* __restrict__ h2, const float* __restrict__ qf32,
    const float* __restrict__ kf32, unsigned short* __restrict__ qbf,
    unsigned short* __restrict__ kbf, unsigned short* __restrict__ hbf) {
  const int lane = threadIdx.x & 31;
  const int wid  = threadIdx.x >> 5;
  const size_t row = (size_t)blockIdx.x * 8 + wid;
  const int y = blockIdx.y;

  const float* src;
  unsigned short* dst;
  bool nrm = true;
  if (y < 3) {
    src = qf32 + (size_t)y * BNH_;       dst = qbf + (size_t)y * BNH_;
  } else if (y < 9) {
    src = kf32 + (size_t)(y - 3) * BNH_; dst = kbf + (size_t)(y - 3) * BNH_;
  } else {
    int i = y - 9;
    src = (i == 0) ? h0 : (i == 1) ? h1 : h2;
    dst = hbf + (size_t)i * BNH_;
    nrm = false;
  }

  float a = src[row * 64 + lane];
  float b = src[row * 64 + 32 + lane];
  float inv = 1.0f;
  if (nrm) {
    float ss = a * a + b * b;
    ss += __shfl_xor(ss, 1);
    ss += __shfl_xor(ss, 2);
    ss += __shfl_xor(ss, 4);
    ss += __shfl_xor(ss, 8);
    ss += __shfl_xor(ss, 16);
    inv = 1.0f / sqrtf(ss);
  }
  dst[row * 64 + lane]      = f2bf(a * inv);
  dst[row * 64 + 32 + lane] = f2bf(b * inv);
}

// ---------------------------------------------------------------------------
// Kernel 3: fused masked cosine attention, single pass over each mask.
// grid (N/64, B, 3), block 128 (4 waves, each one 16-query tile).
// v-tiles are staged global->LDS with double-buffered async copies
// (GLOBAL_LOAD_ASYNC_TO_LDS_B128, ASYNCcnt) overlapping the WMMAs.
// ---------------------------------------------------------------------------
__global__ __launch_bounds__(128) void attn_kernel(
    const unsigned short* __restrict__ qbf,
    const unsigned short* __restrict__ kbf,
    const unsigned short* __restrict__ hbf,
    const float* __restrict__ m0, const float* __restrict__ m1,
    const float* __restrict__ m2, const float* __restrict__ m3,
    const float* __restrict__ m4, const float* __restrict__ m5,
    float* __restrict__ accbuf) {
  // per-wave S staging: 16 rows x 32 bf16, padded to 40 (80B, 16B-aligned,
  // conflict-free for the b128 re-layout reads)
  __shared__ __align__(16) unsigned short slds[4][16][40];
  // per-wave double-buffered v tile: 2 x (32 keys x 64 cols) bf16 = 2 x 4KB
  __shared__ __align__(16) unsigned short vtile[4][2][2048];

  const int lane = threadIdx.x & 31;
  const int wid  = threadIdx.x >> 5;
  const int hf   = lane >> 4;
  const int ln   = lane & 15;
  const int t    = blockIdx.z;
  const int b    = blockIdx.y;
  const int qtile = blockIdx.x * 4 + wid;        // [0, 256)
  const int qbase = b * N_ + qtile * 16;         // row in [0, BN)

  // Preload qn A-fragments (16 rows x K=64 -> 2 fragments, vector loads)
  Frag qa[2];
  const unsigned short* qrow =
      qbf + (size_t)t * BNH_ + (size_t)(qbase + ln) * 64;
#pragma unroll
  for (int f = 0; f < 2; ++f) {
    qa[f].q[0] = *(const u32x4*)(qrow + f * 32 + 8 * hf);
    qa[f].q[1] = *(const u32x4*)(qrow + f * 32 + 16 + 8 * hf);
  }

  v8f accF[4];
#pragma unroll
  for (int nt = 0; nt < 4; ++nt)
    accF[nt] = (v8f){0.f, 0.f, 0.f, 0.f, 0.f, 0.f, 0.f, 0.f};

  for (int j = 0; j < 2; ++j) {
    const int e = t * 2 + j;
    const int pred = edge_pred(e);
    const float* mp = (e == 0) ? m0 : (e == 1) ? m1 : (e == 2) ? m2
                    : (e == 3) ? m3 : (e == 4) ? m4 : m5;
    const unsigned short* kn = kbf + (size_t)e * BNH_ + (size_t)b * N_ * 64;
    const unsigned short* vv = hbf + (size_t)pred * BNH_ + (size_t)b * N_ * 64;

    // async-stage one 32x64 bf16 v chunk (4KB) into LDS buffer `buf`
    auto stage = [&](int buf, int kb) {
      unsigned lds0 =
          (unsigned)(unsigned long long)(&vtile[wid][buf][0]) + lane * 16u;
      unsigned long long ga =
          (unsigned long long)(vv + (size_t)kb * 64) + (unsigned)(lane * 16);
#pragma unroll
      for (int i = 0; i < 8; ++i) {
        asm volatile("global_load_async_to_lds_b128 %0, %1, off"
                     :: "v"(lds0 + i * 512u), "v"(ga + (unsigned)(i * 512))
                     : "memory");
      }
    };

    v8f accE[4];
#pragma unroll
    for (int nt = 0; nt < 4; ++nt)
      accE[nt] = (v8f){0.f, 0.f, 0.f, 0.f, 0.f, 0.f, 0.f, 0.f};
    float rs[8] = {0.f, 0.f, 0.f, 0.f, 0.f, 0.f, 0.f, 0.f};

    stage(0, 0);  // prologue: prefetch chunk 0

    for (int kb = 0; kb < N_; kb += 32) {
      const int buf = (kb >> 5) & 1;
      const bool more = (kb + 32) < N_;
      if (more) stage(buf ^ 1, kb + 32);   // prefetch next chunk

      // --- QK^T: two 16-key subtiles, masked, staged to LDS as bf16 ---
#pragma unroll
      for (int kk = 0; kk < 2; ++kk) {
        const unsigned short* kr = kn + (size_t)(kb + kk * 16 + ln) * 64;
        v8f s = {0.f, 0.f, 0.f, 0.f, 0.f, 0.f, 0.f, 0.f};
#pragma unroll
        for (int f = 0; f < 2; ++f) {
          Frag bk;  // B[h, key] = kn[key, h] : contiguous in h
          bk.q[0] = *(const u32x4*)(kr + f * 32 + 8 * hf);
          bk.q[1] = *(const u32x4*)(kr + f * 32 + 16 + 8 * hf);
          s = wmma_bf16(qa[f], bk, s);
        }
#pragma unroll
        for (int r = 0; r < 8; ++r) {
          int m = r + 8 * hf;
          float mval = mp[(size_t)(qtile * 16 + m) * N_ + kb + kk * 16 + ln];
          rs[r] += mval;                        // rowsum fused into same pass
          slds[wid][m][kk * 16 + ln] = f2bf(s[r] * mval);
        }
      }
      // --- re-layout S (16x32) as an A fragment (same-wave LDS, in-order) ---
      Frag sa;
      {
        const unsigned short* sr = &slds[wid][ln][0];
        sa.q[0] = *(const u32x4*)(sr + 8 * hf);
        sa.q[1] = *(const u32x4*)(sr + 16 + 8 * hf);
      }

      // wait for this chunk's async v copy (next chunk may stay in flight)
      if (more) asm volatile("s_wait_asynccnt 8" ::: "memory");
      else      asm volatile("s_wait_asynccnt 0" ::: "memory");

      // --- AV: 4 column tiles of H=64; B fragments gathered from LDS ---
      const unsigned short* vt = &vtile[wid][buf][0];
#pragma unroll
      for (int nt = 0; nt < 4; ++nt) {
        Frag bv;  // B[k, hcol] = vtile[k*64 + nt*16 + ln]
#pragma unroll
        for (int i = 0; i < 16; ++i)
          bv.u[i] = vt[kmap(i, hf) * 64 + nt * 16 + ln];
        accE[nt] = wmma_bf16(sa, bv, accE[nt]);
      }
    }

    // reduce rowsums across the 16 lanes holding each row, apply 1/rowsum
#pragma unroll
    for (int r = 0; r < 8; ++r) {
      float v = rs[r];
      v += __shfl_xor(v, 1);
      v += __shfl_xor(v, 2);
      v += __shfl_xor(v, 4);
      v += __shfl_xor(v, 8);
      float sc = 1.0f / v;
#pragma unroll
      for (int nt = 0; nt < 4; ++nt) accF[nt][r] += sc * accE[nt][r];
    }
  }

  float* out = accbuf + (size_t)t * BNH_;
#pragma unroll
  for (int nt = 0; nt < 4; ++nt)
#pragma unroll
    for (int r = 0; r < 8; ++r) {
      int m = r + 8 * hf;
      out[(size_t)(qbase + m) * 64 + nt * 16 + ln] = accF[nt][r];
    }
}

// ---------------------------------------------------------------------------
// Kernel 4: gated Euler update + output slice. grid (BN, 3), block 64.
// ---------------------------------------------------------------------------
__global__ __launch_bounds__(64) void update_kernel(
    const float* __restrict__ x0, const float* __restrict__ h0,
    const float* __restrict__ h1, const float* __restrict__ h2,
    const float* __restrict__ acc, const float* __restrict__ Wsu0,
    const float* __restrict__ Wsu12, const float* __restrict__ Wdt,
    const float* __restrict__ bdt, const float* __restrict__ step_size,
    const float* __restrict__ out_scale, float* __restrict__ out) {
  const int o = threadIdx.x;        // 0..63
  const size_t row = blockIdx.x;    // 0..BN-1
  const int t = blockIdx.y;

  const float* h = (t == 0) ? h0 : (t == 1) ? h1 : h2;
  const float* hr = h + row * 64;

  float z = bdt[t * 64 + o];
  const float* wd = Wdt + (size_t)t * 4096 + (size_t)o * 64;
  for (int k = 0; k < 64; ++k) z += hr[k] * wd[k];
  float dt = 2.0f / (1.0f + __expf(-z)) - 1.0f;

  float u = 0.f;
  const float* ar = acc + (size_t)t * BNH_ + row * 64;
  if (t == 0) {
    const float* xr = x0 + row * 32;
    const float* w = Wsu0 + (size_t)o * 96;
    for (int k = 0; k < 32; ++k) u += xr[k] * w[k];
    for (int k = 0; k < 64; ++k) u += ar[k] * w[32 + k];
  } else {
    const float* w = Wsu12 + (size_t)(t - 1) * 4096 + (size_t)o * 64;
    for (int k = 0; k < 64; ++k) u += ar[k] * w[k];
  }

  float hn = hr[o] + step_size[0] * dt * u;
  out[OUTOFF_ + (size_t)t * BNH_ + row * 64 + o] = hn;
  if (t == 2 && o < 8) out[row * 8 + o] = out_scale[0] * hn;
}

// ---------------------------------------------------------------------------
extern "C" void kernel_launch(void* const* d_in, const int* in_sizes, int n_in,
                              void* d_out, int out_size, void* d_ws,
                              size_t ws_size, hipStream_t stream) {
  const float* x0 = (const float*)d_in[0];
  const float* h0 = (const float*)d_in[1];
  const float* h1 = (const float*)d_in[2];
  const float* h2 = (const float*)d_in[3];
  const float* m0 = (const float*)d_in[4];   // m00
  const float* m1 = (const float*)d_in[5];   // m20
  const float* m2 = (const float*)d_in[6];   // m01
  const float* m3 = (const float*)d_in[7];   // m11
  const float* m4 = (const float*)d_in[8];   // m12
  const float* m5 = (const float*)d_in[9];   // m22
  const float* Wq    = (const float*)d_in[10];
  const float* Wk    = (const float*)d_in[11];
  const float* Wsu0  = (const float*)d_in[12];
  const float* Wsu12 = (const float*)d_in[13];
  const float* Wdt   = (const float*)d_in[14];
  const float* bdt   = (const float*)d_in[15];
  const float* step_size = (const float*)d_in[16];
  const float* out_scale = (const float*)d_in[17];
  float* out = (float*)d_out;

  // workspace layout (bytes): total ~37.75 MB
  char* base = (char*)d_ws;
  float*          qf32 = (float*)(base);                              // 3*BNH f32
  float*          kf32 = (float*)(base + 6291456);                    // 6*BNH f32
  unsigned short* qbf  = (unsigned short*)(base + 18874368);          // 3*BNH bf16
  unsigned short* kbf  = (unsigned short*)(base + 22020096);          // 6*BNH bf16
  unsigned short* hbf  = (unsigned short*)(base + 28311552);          // 3*BNH bf16
  float*          acc  = (float*)(base + 31457280);                   // 3*BNH f32

  proj_kernel<<<dim3(128, 9), 128, 0, stream>>>(h0, h1, h2, Wq, Wk, qf32, kf32);
  norm_kernel<<<dim3(BN_ / 8, 12), 256, 0, stream>>>(h0, h1, h2, qf32, kf32,
                                                     qbf, kbf, hbf);
  attn_kernel<<<dim3(N_ / 64, B_, 3), 128, 0, stream>>>(qbf, kbf, hbf, m0, m1,
                                                        m2, m3, m4, m5, acc);
  update_kernel<<<dim3(BN_, 3), 64, 0, stream>>>(x0, h0, h1, h2, acc, Wsu0,
                                                 Wsu12, Wdt, bdt, step_size,
                                                 out_scale, out);
}
